// PureTorchMambaBlock_68917045231758
// MI455X (gfx1250) — matrix operand order repaired and checked by reading, not compiled, not run on hardware
//
#include <hip/hip_runtime.h>
#include <hip/hip_bf16.h>

// ---------------------------------------------------------------------------
// Mamba block forward for gfx1250 (MI455X).
// GEMM1/GEMM3: v_wmma_f32_16x16x32_f16, 32x64 wave tile, copy-free 2-stage
// software pipeline, __launch_bounds__ to avoid spills.
// Selective scan: 3-pass chunked linear-recurrence scan (32 chunks x 64).
// ---------------------------------------------------------------------------

typedef _Float16 v8h  __attribute__((ext_vector_type(8)));
typedef _Float16 v16h __attribute__((ext_vector_type(16)));
typedef float    v8f  __attribute__((ext_vector_type(8)));

#define D_MODEL 1024
#define D_STATE 16
#define D_CONV  4
#define D_INNER 2048
#define BATCH   2
#define SEQ     2048
#define TOK     (BATCH * SEQ)        // 4096
#define NCHUNK  32
#define CLEN    (SEQ / NCHUNK)       // 64
#define LN_EPS  1e-5f

// ---------------------------------------------------------------- helpers ---
__device__ __forceinline__ float sigmoid_f(float v) {
  return 1.0f / (1.0f + __expf(-v));
}
__device__ __forceinline__ float silu_f(float v) { return v * sigmoid_f(v); }
__device__ __forceinline__ float softplus_f(float v) {
  return (v > 20.0f) ? v : log1pf(__expf(v));
}

// ------------------------------------------------------------- layernorm ---
__global__ void ln_to_f16_kernel(const float* __restrict__ x,
                                 const float* __restrict__ w,
                                 const float* __restrict__ b,
                                 _Float16* __restrict__ out) {
  __shared__ float s1[256];
  __shared__ float s2[256];
  const int row = blockIdx.x;
  const int tid = threadIdx.x;
  const float* xr = x + (size_t)row * D_MODEL;
  float s = 0.f, q = 0.f;
  for (int i = tid; i < D_MODEL; i += 256) {
    float v = xr[i];
    s += v; q += v * v;
  }
  s1[tid] = s; s2[tid] = q;
  __syncthreads();
  for (int off = 128; off > 0; off >>= 1) {
    if (tid < off) { s1[tid] += s1[tid + off]; s2[tid] += s2[tid + off]; }
    __syncthreads();
  }
  const float mu  = s1[0] * (1.0f / D_MODEL);
  const float var = s2[0] * (1.0f / D_MODEL) - mu * mu;
  const float rs  = rsqrtf(var + LN_EPS);
  _Float16* orow = out + (size_t)row * D_MODEL;
  for (int i = tid; i < D_MODEL; i += 256) {
    orow[i] = (_Float16)((xr[i] - mu) * rs * w[i] + b[i]);
  }
}

// ------------------------------------------------------------ f32 -> f16 ---
__global__ void cvt_f16_kernel(const float* __restrict__ in,
                               _Float16* __restrict__ out, int n) {
  int i = blockIdx.x * 256 + threadIdx.x;
  if (i < n) out[i] = (_Float16)in[i];
}

// ------------------------------------------------------------- WMMA GEMM ---
// Out[M,N] = A[M,K] * W[N,K]^T  (+Res), A/W f16 K-contiguous, Out f32.
// Wave tile 32(M) x 64(N): 2 A-frags x 4 B-frags, 8 WMMAs per K-step.
// Block = 8 waves arranged 4(M) x 2(N) -> 128x128 block tile.
// Requires K % 64 == 0 (even number of 32-wide K-steps).
__device__ __forceinline__ v16h load_a_frag(const _Float16* __restrict__ A,
                                            int lda, int m0, int k0, int lane) {
  const int r = lane & 15, hi = lane >> 4;
  const _Float16* base = A + (size_t)(m0 + r) * lda + (k0 + hi * 16);
  return *(const v16h*)(base);         // K = k0 + hi*16 .. +15
}
__device__ __forceinline__ v16h load_b_frag(const _Float16* __restrict__ W,
                                            int ldb, int n0, int k0, int lane) {
  const int c = lane & 15, hi = lane >> 4;
  return *(const v16h*)(W + (size_t)(n0 + c) * ldb + (k0 + hi * 16));
}

struct Frags {
  v16h a[2];
  v16h b[4];
};

__device__ __forceinline__ void load_frags(Frags& f,
                                           const _Float16* __restrict__ A,
                                           const _Float16* __restrict__ W,
                                           int K, int m0, int n0, int k0,
                                           int lane) {
#pragma unroll
  for (int mi = 0; mi < 2; ++mi) f.a[mi] = load_a_frag(A, K, m0 + 16 * mi, k0, lane);
#pragma unroll
  for (int ni = 0; ni < 4; ++ni) f.b[ni] = load_b_frag(W, K, n0 + 16 * ni, k0, lane);
}

__device__ __forceinline__ void mma_step(const Frags& f, v8f (&acc)[2][4]) {
#pragma unroll
  for (int mi = 0; mi < 2; ++mi)
#pragma unroll
    for (int ni = 0; ni < 4; ++ni)
      acc[mi][ni] = __builtin_amdgcn_wmma_f32_16x16x32_f16(
          false, f.a[mi], false, f.b[ni], (short)0, acc[mi][ni], false, false);
}

template <bool RES>
__global__ __launch_bounds__(256, 1)
void wmma_gemm_kernel(const _Float16* __restrict__ A,
                      const _Float16* __restrict__ W,
                      const float* __restrict__ Res,
                      float* __restrict__ Out,
                      int M, int N, int K) {
  const int lane = threadIdx.x & 31;
  const int wave = threadIdx.x >> 5;          // 0..7
  const int wm = wave >> 1;                   // 0..3
  const int wn = wave & 1;                    // 0..1
  const int m0 = blockIdx.y * 128 + wm * 32;
  const int n0 = blockIdx.x * 128 + wn * 64;

  v8f acc[2][4] = {};

  Frags f0, f1;
  load_frags(f0, A, W, K, m0, n0, 0, lane);
  load_frags(f1, A, W, K, m0, n0, 32, lane);

  for (int k0 = 0; k0 < K; k0 += 64) {
    mma_step(f0, acc);
    if (k0 + 64 < K) {
      __builtin_prefetch(A + (size_t)(m0 + (lane & 15)) * K + (k0 + 128), 0, 1);
      load_frags(f0, A, W, K, m0, n0, k0 + 64, lane);   // reload in place
    }
    mma_step(f1, acc);
    if (k0 + 96 < K) {
      __builtin_prefetch(W + (size_t)(n0 + (lane & 15)) * K + (k0 + 128), 0, 1);
      load_frags(f1, A, W, K, m0, n0, k0 + 96, lane);   // reload in place
    }
  }

  // C/D layout: lane c = N, VGPR r holds M = r + 8*(lane>=16)
  const int hi = lane >> 4, c = lane & 15;
#pragma unroll
  for (int mi = 0; mi < 2; ++mi)
#pragma unroll
    for (int ni = 0; ni < 4; ++ni)
#pragma unroll
      for (int r = 0; r < 8; ++r) {
        const int row = m0 + mi * 16 + 8 * hi + r;
        const int col = n0 + ni * 16 + c;
        const size_t o = (size_t)row * N + col;
        float v = acc[mi][ni][r];
        if (RES) v += Res[o];
        Out[o] = v;
      }
}

// ----------------------------------------------- depthwise conv4 + SiLU ---
// x_p = xz[:, :D_INNER]; xc[b,t,d] = silu(sum_i x_p[b,t-3+i,d]*cw[d,i] + cb[d])
__global__ void conv_silu_kernel(const float* __restrict__ xz,
                                 const float* __restrict__ cw,
                                 const float* __restrict__ cb,
                                 float* __restrict__ xc) {
  const size_t idx = (size_t)blockIdx.x * 256 + threadIdx.x;  // b*2^22+t*2^11+d
  const int d = (int)(idx & (D_INNER - 1));
  const int t = (int)((idx >> 11) & (SEQ - 1));
  const int b = (int)(idx >> 22);
  float acc = cb[d];
#pragma unroll
  for (int i = 0; i < D_CONV; ++i) {
    const int tt = t - (D_CONV - 1) + i;
    if (tt >= 0)
      acc = fmaf(xz[(size_t)(b * SEQ + tt) * (2 * D_INNER) + d], cw[d * D_CONV + i], acc);
  }
  xc[idx] = silu_f(acc);
}

// -------------------------------------------------- dbc = xc @ W_x^T (33) ---
__global__ void dbc_kernel(const float* __restrict__ xc,
                           const float* __restrict__ Wx,
                           float* __restrict__ dbc) {
  __shared__ float srow[D_INNER];
  const int row = blockIdx.x;
  const int tid = threadIdx.x;
  for (int i = tid; i < D_INNER; i += 256)
    srow[i] = xc[(size_t)row * D_INNER + i];
  __syncthreads();
  const int lane = tid & 31;
  const int wave = tid >> 5;
  for (int j = wave; j < 2 * D_STATE + 1; j += 8) {
    const float* wr = Wx + (size_t)j * D_INNER;
    float s = 0.f;
    for (int i = lane; i < D_INNER; i += 32) s = fmaf(srow[i], wr[i], s);
#pragma unroll
    for (int off = 16; off > 0; off >>= 1) s += __shfl_xor(s, off, 32);
    if (lane == 0) dbc[(size_t)row * 33 + j] = s;
  }
}

// --------------------------------------------------- scan pass A: carries ---
__global__ void scan_carry_kernel(const float* __restrict__ dbc,
                                  const float* __restrict__ xc,
                                  const float* __restrict__ A_log,
                                  const float* __restrict__ W_dt,
                                  const float* __restrict__ b_dt,
                                  float* __restrict__ carryA,
                                  float* __restrict__ carryH) {
  const int gid  = blockIdx.x;              // b*(NCHUNK*8) + c*8 + dblk
  const int dblk = gid & 7;
  const int c    = (gid >> 3) & (NCHUNK - 1);
  const int b    = gid >> 8;
  const int d    = dblk * 256 + threadIdx.x;

  float a[D_STATE], h[D_STATE], p[D_STATE];
  const float wdt = W_dt[d], bdt = b_dt[d];
#pragma unroll
  for (int n = 0; n < D_STATE; ++n) {
    a[n] = -__expf(A_log[(size_t)d * D_STATE + n]);
    h[n] = 0.f; p[n] = 1.f;
  }
  const int t0 = c * CLEN;
  for (int t = t0; t < t0 + CLEN; ++t) {
    const int row = b * SEQ + t;
    const float* dr = dbc + (size_t)row * 33;
    const float dtraw = dr[0];
    const float xcv = xc[(size_t)row * D_INNER + d];
    const float dt  = softplus_f(fmaf(dtraw, wdt, bdt));
    const float dtx = dt * xcv;
#pragma unroll
    for (int n = 0; n < D_STATE; ++n) {
      const float e = __expf(dt * a[n]);
      h[n] = fmaf(e, h[n], dtx * dr[1 + n]);
      p[n] *= e;
    }
  }
  const size_t base = ((size_t)(b * NCHUNK + c) * D_INNER + d) * D_STATE;
#pragma unroll
  for (int n = 0; n < D_STATE; ++n) {
    carryA[base + n] = p[n];
    carryH[base + n] = h[n];
  }
}

// ------------------------------------------- scan pass B: combine chunks ---
__global__ void scan_combine_kernel(const float* __restrict__ carryA,
                                    const float* __restrict__ carryH,
                                    float* __restrict__ hin) {
  const int idx = blockIdx.x * 256 + threadIdx.x;   // (b,d,n): 65536 total
  const int b = idx >> 15;
  const int dn = idx & 32767;                        // d*16 + n
  float h = 0.f;
#pragma unroll 1
  for (int c = 0; c < NCHUNK; ++c) {
    const size_t o = (size_t)(b * NCHUNK + c) * (D_INNER * D_STATE) + dn;
    hin[o] = h;
    h = fmaf(carryA[o], h, carryH[o]);
  }
}

// ---------------------- scan pass C: replay with carry-in + fused epilogue --
__global__ void scan_apply_kernel(const float* __restrict__ dbc,
                                  const float* __restrict__ xc,
                                  const float* __restrict__ xz,
                                  const float* __restrict__ A_log,
                                  const float* __restrict__ W_dt,
                                  const float* __restrict__ b_dt,
                                  const float* __restrict__ Dp,
                                  const float* __restrict__ hin,
                                  _Float16* __restrict__ y_h) {
  const int gid  = blockIdx.x;
  const int dblk = gid & 7;
  const int c    = (gid >> 3) & (NCHUNK - 1);
  const int b    = gid >> 8;
  const int d    = dblk * 256 + threadIdx.x;

  float a[D_STATE], h[D_STATE];
  const float wdt = W_dt[d], bdt = b_dt[d], Dv = Dp[d];
  const size_t base = ((size_t)(b * NCHUNK + c) * D_INNER + d) * D_STATE;
#pragma unroll
  for (int n = 0; n < D_STATE; ++n) {
    a[n] = -__expf(A_log[(size_t)d * D_STATE + n]);
    h[n] = hin[base + n];
  }
  const int t0 = c * CLEN;
  for (int t = t0; t < t0 + CLEN; ++t) {
    const int row = b * SEQ + t;
    const float* dr = dbc + (size_t)row * 33;
    const float dtraw = dr[0];
    const float xcv = xc[(size_t)row * D_INNER + d];
    const float dt  = softplus_f(fmaf(dtraw, wdt, bdt));
    const float dtx = dt * xcv;
    float y = 0.f;
#pragma unroll
    for (int n = 0; n < D_STATE; ++n) {
      const float e = __expf(dt * a[n]);
      h[n] = fmaf(e, h[n], dtx * dr[1 + n]);
      y = fmaf(h[n], dr[17 + n], y);
    }
    const float z = xz[(size_t)row * (2 * D_INNER) + D_INNER + d];
    const float out = (y + xcv * Dv) * silu_f(z);
    y_h[(size_t)row * D_INNER + d] = (_Float16)out;
  }
}

// ------------------------------------------------------------------ entry ---
extern "C" void kernel_launch(void* const* d_in, const int* in_sizes, int n_in,
                              void* d_out, int out_size, void* d_ws, size_t ws_size,
                              hipStream_t stream) {
  const float* x      = (const float*)d_in[0];
  const float* ln_w   = (const float*)d_in[1];
  const float* ln_b   = (const float*)d_in[2];
  const float* W_in   = (const float*)d_in[3];
  const float* conv_w = (const float*)d_in[4];
  const float* conv_b = (const float*)d_in[5];
  const float* W_x    = (const float*)d_in[6];
  const float* W_dt   = (const float*)d_in[7];
  const float* b_dt   = (const float*)d_in[8];
  const float* A_log  = (const float*)d_in[9];
  const float* Dp     = (const float*)d_in[10];
  const float* W_out  = (const float*)d_in[11];
  float* out = (float*)d_out;

  // ---- workspace layout -------------------------------------------------
  size_t off = 0;
  auto alloc = [&](size_t bytes) -> void* {
    void* p = (char*)d_ws + off;
    off = (off + bytes + 255) & ~(size_t)255;
    return p;
  };
  _Float16* xn_h   = (_Float16*)alloc((size_t)TOK * D_MODEL * 2);
  _Float16* Win_h  = (_Float16*)alloc((size_t)2 * D_INNER * D_MODEL * 2);
  _Float16* Wout_h = (_Float16*)alloc((size_t)D_MODEL * D_INNER * 2);
  float*    xz     = (float*)alloc((size_t)TOK * 2 * D_INNER * 4);
  float*    xc     = (float*)alloc((size_t)TOK * D_INNER * 4);
  float*    dbc    = (float*)alloc((size_t)TOK * 33 * 4);
  float*    carryA = (float*)alloc((size_t)BATCH * NCHUNK * D_INNER * D_STATE * 4);
  float*    carryH = (float*)alloc((size_t)BATCH * NCHUNK * D_INNER * D_STATE * 4);
  float*    hin    = (float*)alloc((size_t)BATCH * NCHUNK * D_INNER * D_STATE * 4);
  _Float16* y_h    = (_Float16*)alloc((size_t)TOK * D_INNER * 2);
  (void)ws_size; (void)in_sizes; (void)n_in; (void)out_size;

  // 1) LayerNorm -> f16 A for GEMM1
  ln_to_f16_kernel<<<TOK, 256, 0, stream>>>(x, ln_w, ln_b, xn_h);

  // 2) Weight conversions (per call; no cached state)
  {
    int n1 = 2 * D_INNER * D_MODEL;
    cvt_f16_kernel<<<(n1 + 255) / 256, 256, 0, stream>>>(W_in, Win_h, n1);
    int n2 = D_MODEL * D_INNER;
    cvt_f16_kernel<<<(n2 + 255) / 256, 256, 0, stream>>>(W_out, Wout_h, n2);
  }

  // 3) GEMM1: xz[4096,4096] = xn[4096,1024] @ W_in^T   (WMMA, 128x128 blocks)
  {
    dim3 grid((2 * D_INNER) / 128, TOK / 128);
    wmma_gemm_kernel<false><<<grid, 256, 0, stream>>>(
        xn_h, Win_h, nullptr, xz, TOK, 2 * D_INNER, D_MODEL);
  }

  // 4) depthwise conv + SiLU
  conv_silu_kernel<<<(size_t)TOK * D_INNER / 256, 256, 0, stream>>>(xz, conv_w, conv_b, xc);

  // 5) dbc = xc @ W_x^T  (N=33)
  dbc_kernel<<<TOK, 256, 0, stream>>>(xc, W_x, dbc);

  // 6) chunked selective scan
  scan_carry_kernel<<<BATCH * NCHUNK * (D_INNER / 256), 256, 0, stream>>>(
      dbc, xc, A_log, W_dt, b_dt, carryA, carryH);
  scan_combine_kernel<<<(BATCH * D_INNER * D_STATE) / 256, 256, 0, stream>>>(
      carryA, carryH, hin);
  scan_apply_kernel<<<BATCH * NCHUNK * (D_INNER / 256), 256, 0, stream>>>(
      dbc, xc, xz, A_log, W_dt, b_dt, Dp, hin, y_h);

  // 7) GEMM3: out = y @ W_out^T + residual   (WMMA, 128x128 blocks)
  {
    dim3 grid(D_MODEL / 128, TOK / 128);
    wmma_gemm_kernel<true><<<grid, 256, 0, stream>>>(
        y_h, Wout_h, x, out, TOK, D_MODEL, D_INNER);
  }
}
